// CompExtractor_78056735638032
// MI455X (gfx1250) — compile-verified, matrix operand order checked by reading, not run on hardware
//
#include <hip/hip_runtime.h>
#include <hip/hip_bf16.h>

typedef __bf16 bf16_t;
typedef __attribute__((ext_vector_type(16))) __bf16 v16bf;
typedef __attribute__((ext_vector_type(8)))  float  v8f;

struct alignas(16) I4 { int x, y, z, w; };
union ABu { v16bf v; I4 q[2]; };

#define GB  64
#define GN  512
#define GNBS 8
#define GE  2048
#define GH  128
#define GL  3
#define GK  4
#define GM  (GB*GN)          // 32768 vertex rows
#define ALPHA_C 0.1f
#define LAMDA_C 0.5f

__device__ __forceinline__ float gelu_f(float x) {
  return 0.5f * x * (1.f + erff(x * 0.70710678118654752f));
}
__device__ __forceinline__ float sigm_f(float x) {
  return 1.f / (1.f + __expf(-x));
}
__device__ __forceinline__ void ldB(ABu& b, const bf16_t* p) {
  b.q[0] = *(const I4*)(p);
  b.q[1] = *(const I4*)(p + 16);
}

// ---------------------------------------------------------------------------
// Generic bf16 WMMA GEMM: out = epilogue(A[M][K] @ Wt[Ncol][K] + bias)
// block = 256 threads = 8 waves; wave -> 16 rows; block -> 128 rows x 128 cols
// - Weight tile staged in LDS (row stride K+8 bf16 -> conflict-free b128).
// - All NS=K/32 A fragments loaded up-front as one clause (SSA, no copies).
// - B operands 2-deep software-pipelined from LDS (wait dscnt<=2 per WMMA).
// Epilogue: v = acc + bias[col]; if pre: v += pre[(m>>preShift)*128+col];
//           act 0 none / 1 gelu / 2 sigmoid / 3 tanh;
//           mode 0: v = alpha*v + beta*post[m*ldc+col]      (if post)
//           mode 1: v = (1-v)*post[m*ldc+col] + v*bc2[(m>>preShift)*128+col]
// dynamic LDS bytes = 128*(K+8)*2
// ---------------------------------------------------------------------------
template<int K>
__global__ __launch_bounds__(256)
void k_gemm(const bf16_t* __restrict__ A, int lda,
            const bf16_t* __restrict__ Wt, int ldw,
            const float* __restrict__ bias,
            float* __restrict__ out32, bf16_t* __restrict__ out16, int ldc,
            const float* __restrict__ pre, int preShift,
            int act, int mode,
            const float* __restrict__ post, const float* __restrict__ bc2,
            float alpha, float beta)
{
  extern __shared__ bf16_t lwt[];            // [128][K+8]
  constexpr int NS = K / 32;
  constexpr int lstr = K + 8;                // LDS row stride (bf16 elems)
  const int lane = threadIdx.x & 31;
  const int wave = threadIdx.x >> 5;
  const int hl   = lane >> 4;                // lane-half: selects K sub-chunk
  const int l16  = lane & 15;
  const int row0 = blockIdx.x * 128 + wave * 16;
  const int col0 = blockIdx.y * 128;

  // cooperative stage of the 128 x K weight tile into LDS (16B chunks)
  constexpr int CH = K >> 3;
  for (int idx = threadIdx.x; idx < 128 * CH; idx += 256) {
    const int row = idx / CH, c = idx % CH;
    *(I4*)(lwt + row * lstr + c * 8) =
        *(const I4*)(Wt + (size_t)(col0 + row) * ldw + c * 8);
  }

  // load the wave's complete A fragment set up-front (one clause)
  const bf16_t* Ap = A + (size_t)(row0 + l16) * lda + hl * 8;
  ABu a[NS];
#pragma unroll
  for (int s = 0; s < NS; ++s) {
    a[s].q[0] = *(const I4*)(Ap + s * 32);
    a[s].q[1] = *(const I4*)(Ap + s * 32 + 16);
  }
  // prefetch the epilogue's additive operand rows while MACs run
  if (post) __builtin_prefetch(post + (size_t)(row0 + hl * 8) * ldc + col0, 0, 1);
  __syncthreads();

  const v8f zero = {0.f,0.f,0.f,0.f,0.f,0.f,0.f,0.f};
  v8f acc[8];
#pragma unroll
  for (int i = 0; i < 8; ++i) acc[i] = zero;

#pragma unroll
  for (int s = 0; s < NS; ++s) {
    const int kk = s * 32;
    const bf16_t* Bp = lwt + l16 * lstr + hl * 8 + kk;
    ABu b0, b1;
    ldB(b0, Bp);
#pragma unroll
    for (int ct = 0; ct < 8; ct += 2) {
      ldB(b1, Bp + (ct + 1) * 16 * lstr);
      acc[ct] = __builtin_amdgcn_wmma_f32_16x16x32_bf16(
          false, a[s].v, false, b0.v, (short)0, acc[ct], false, false);
      if (ct + 2 < 8) ldB(b0, Bp + (ct + 2) * 16 * lstr);
      acc[ct + 1] = __builtin_amdgcn_wmma_f32_16x16x32_bf16(
          false, a[s].v, false, b1.v, (short)0, acc[ct + 1], false, false);
    }
  }

#pragma unroll
  for (int ct = 0; ct < 8; ++ct) {
    const int col = col0 + ct * 16 + l16;
    const float bi = bias ? bias[col] : 0.f;
#pragma unroll
    for (int r = 0; r < 8; ++r) {
      const int m = row0 + hl * 8 + r;       // D layout: lanes16-31 hold M = r+8
      float v = acc[ct][r] + bi;
      if (pre) v += pre[((size_t)(m >> preShift)) * 128 + col];
      if (act == 1)      v = gelu_f(v);
      else if (act == 2) v = sigm_f(v);
      else if (act == 3) v = tanhf(v);
      if (mode == 1)
        v = (1.f - v) * post[(size_t)m * ldc + col] +
            v * bc2[((size_t)(m >> preShift)) * 128 + col];
      else if (post)
        v = alpha * v + beta * post[(size_t)m * ldc + col];
      if (out32) out32[(size_t)m * ldc + col] = v;
      if (out16) out16[(size_t)m * ldc + col] = (bf16_t)v;
    }
  }
}

// ---------------------------------------------------------------------------
// score[b,k,n] = sum_d tanh((vf @ Wvm_k)[n,d] + bvm[k,d]) * master[b,d]*wbmm[k,d]
//               + wbmm_b[k]
// grid = (N/128, K, B).  WMMA GEMM (LDS-staged W) + tanh + lane row-reduce.
// ---------------------------------------------------------------------------
__global__ __launch_bounds__(256)
void k_score(const bf16_t* __restrict__ vf16,
             const bf16_t* __restrict__ Wtv,     // [K][128][128] bf16 (layer)
             const float* __restrict__ wvm_b,    // [K][128] (layer)
             const float* __restrict__ master,   // [B][128]
             const float* __restrict__ wbmm_w,   // [K][128] (layer)
             const float* __restrict__ wbmm_b,   // [K] (layer)
             float* __restrict__ score)          // [B][K][N]
{
  __shared__ bf16_t lwt[128 * 136];              // [128][128+8]
  const int lane = threadIdx.x & 31, wave = threadIdx.x >> 5;
  const int hl = lane >> 4, l16 = lane & 15;
  const int k = blockIdx.y, b = blockIdx.z;
  const int row0 = blockIdx.x * 128 + wave * 16;

  const bf16_t* Wk = Wtv + (size_t)k * GH * GH;
  for (int idx = threadIdx.x; idx < 128 * 16; idx += 256) {
    const int row = idx >> 4, c = idx & 15;
    *(I4*)(lwt + row * 136 + c * 8) = *(const I4*)(Wk + (size_t)row * GH + c * 8);
  }

  const bf16_t* Ap = vf16 + ((size_t)b * GN + row0 + l16) * GH + hl * 8;
  ABu a[4];
#pragma unroll
  for (int s = 0; s < 4; ++s) {
    a[s].q[0] = *(const I4*)(Ap + s * 32);
    a[s].q[1] = *(const I4*)(Ap + s * 32 + 16);
  }
  __syncthreads();

  const v8f zero = {0.f,0.f,0.f,0.f,0.f,0.f,0.f,0.f};
  v8f acc[8];
#pragma unroll
  for (int i = 0; i < 8; ++i) acc[i] = zero;

#pragma unroll
  for (int s = 0; s < 4; ++s) {
    const int kk = s * 32;
    const bf16_t* Bp = lwt + l16 * 136 + hl * 8 + kk;
    ABu b0, b1;
    ldB(b0, Bp);
#pragma unroll
    for (int ct = 0; ct < 8; ct += 2) {
      ldB(b1, Bp + (ct + 1) * 16 * 136);
      acc[ct] = __builtin_amdgcn_wmma_f32_16x16x32_bf16(
          false, a[s].v, false, b0.v, (short)0, acc[ct], false, false);
      if (ct + 2 < 8) ldB(b0, Bp + (ct + 2) * 16 * 136);
      acc[ct + 1] = __builtin_amdgcn_wmma_f32_16x16x32_bf16(
          false, a[s].v, false, b1.v, (short)0, acc[ct + 1], false, false);
    }
  }

  float part[8];
#pragma unroll
  for (int r = 0; r < 8; ++r) part[r] = 0.f;
#pragma unroll
  for (int ct = 0; ct < 8; ++ct) {
    const int col = ct * 16 + l16;
    const float bi = wvm_b[k * GH + col];
    const float w  = master[b * GH + col] * wbmm_w[k * GH + col];
#pragma unroll
    for (int r = 0; r < 8; ++r) part[r] += tanhf(acc[ct][r] + bi) * w;
  }
  // reduce over the 16 column-lanes of each half-wave
#pragma unroll
  for (int mmask = 1; mmask < 16; mmask <<= 1)
#pragma unroll
    for (int r = 0; r < 8; ++r) part[r] += __shfl_xor(part[r], mmask, 32);
  if (l16 == 0) {
#pragma unroll
    for (int r = 0; r < 8; ++r)
      score[((size_t)b * GK + k) * GN + row0 + hl * 8 + r] = part[r] + wbmm_b[k];
  }
}

// ---------------------------------------------------------------------------
// Neighbor conv: for each (b,n,j) row m: A row = [vf16[aa[m]] | e16[ba[m]] | 0]
// lab = gelu(A @ u1t + b1); nl[v,col] = sum_j lab * nbs_mask  -> cat2[v][0:128]
// Also copies this block's 16 vertex rows of vf16 into cat2[v][128:256].
// K = 160 (NS=5).  grid.x = 2048.
// ---------------------------------------------------------------------------
__device__ __forceinline__ I4 nbr_chunk(const bf16_t* vf16, const bf16_t* e16,
                                        int av, int bv, int e) {
  if (e < 128) return *(const I4*)(vf16 + (size_t)av * GH + e);
  if (e == 128) return *(const I4*)(e16 + (size_t)bv * 8);
  I4 z; z.x = z.y = z.z = z.w = 0; return z;
}

__global__ __launch_bounds__(256)
void k_nbr(const bf16_t* __restrict__ vf16, const bf16_t* __restrict__ e16,
           const int* __restrict__ aa, const int* __restrict__ ba,
           const float* __restrict__ nm,
           const bf16_t* __restrict__ Wt,      // [128][160] bf16 (layer)
           const float* __restrict__ bias,     // [128] (layer)
           bf16_t* __restrict__ cat2)          // [B*N][256]
{
  __shared__ bf16_t lwt[128 * 168];            // [128][160+8]
  const int lane = threadIdx.x & 31, wave = threadIdx.x >> 5;
  const int hl = lane >> 4, l16 = lane & 15;
  const int row0 = blockIdx.x * 128 + wave * 16;
  const int arow = row0 + l16;
  const int av = aa[arow];
  const int bv = ba[arow];

  for (int idx = threadIdx.x; idx < 128 * 20; idx += 256) {
    const int row = idx / 20, c = idx % 20;
    *(I4*)(lwt + row * 168 + c * 8) = *(const I4*)(Wt + (size_t)row * 160 + c * 8);
  }
  // concat: copy this block's 16 vertex rows of vf16 into cat2 cols 128:256
  {
    const int vbase = blockIdx.x * 16;
    const int idx = threadIdx.x;               // exactly 256 chunks
    const int row = idx >> 4, c = idx & 15;
    *(I4*)(cat2 + (size_t)(vbase + row) * 256 + 128 + c * 8) =
        *(const I4*)(vf16 + (size_t)(vbase + row) * GH + c * 8);
  }

  ABu a[5];
#pragma unroll
  for (int s = 0; s < 5; ++s) {
    a[s].q[0] = nbr_chunk(vf16, e16, av, bv, s * 32 + hl * 8);
    a[s].q[1] = nbr_chunk(vf16, e16, av, bv, s * 32 + hl * 8 + 16);
  }
  __syncthreads();

  const v8f zero = {0.f,0.f,0.f,0.f,0.f,0.f,0.f,0.f};
  v8f acc[8];
#pragma unroll
  for (int i = 0; i < 8; ++i) acc[i] = zero;

#pragma unroll
  for (int s = 0; s < 5; ++s) {
    const int kk = s * 32;
    const bf16_t* Bp = lwt + l16 * 168 + hl * 8 + kk;
    ABu b0, b1;
    ldB(b0, Bp);
#pragma unroll
    for (int ct = 0; ct < 8; ct += 2) {
      ldB(b1, Bp + (ct + 1) * 16 * 168);
      acc[ct] = __builtin_amdgcn_wmma_f32_16x16x32_bf16(
          false, a[s].v, false, b0.v, (short)0, acc[ct], false, false);
      if (ct + 2 < 8) ldB(b0, Bp + (ct + 2) * 16 * 168);
      acc[ct + 1] = __builtin_amdgcn_wmma_f32_16x16x32_bf16(
          false, a[s].v, false, b1.v, (short)0, acc[ct + 1], false, false);
    }
  }

  float nmr[8];
#pragma unroll
  for (int r = 0; r < 8; ++r) nmr[r] = nm[row0 + hl * 8 + r];
  const int vtx = (row0 >> 3) + hl;
#pragma unroll
  for (int ct = 0; ct < 8; ++ct) {
    const int col = ct * 16 + l16;
    const float bi = bias[col];
    float s = 0.f;
#pragma unroll
    for (int r = 0; r < 8; ++r) s += gelu_f(acc[ct][r] + bi) * nmr[r];
    cat2[(size_t)vtx * 256 + col] = (bf16_t)s;
  }
}

// ---------------------------------------------------------------------------
// masked softmax over N=512 per (b,k).  grid = B*K, block = 256.
// ---------------------------------------------------------------------------
__global__ __launch_bounds__(256)
void k_softmax(const float* __restrict__ sc, const float* __restrict__ vm,
               float* __restrict__ att)
{
  const int bk = blockIdx.x, t = threadIdx.x, b = bk >> 2;
  __shared__ float red[256];
  float x0 = sc[(size_t)bk * GN + t];
  float x1 = sc[(size_t)bk * GN + 256 + t];
  red[t] = fmaxf(x0, x1);
  __syncthreads();
  for (int s = 128; s > 0; s >>= 1) {
    if (t < s) red[t] = fmaxf(red[t], red[t + s]);
    __syncthreads();
  }
  const float mx = red[0];
  __syncthreads();
  float e0 = __expf(x0 - mx) * vm[b * GN + t];
  float e1 = __expf(x1 - mx) * vm[b * GN + 256 + t];
  red[t] = e0 + e1;
  __syncthreads();
  for (int s = 128; s > 0; s >>= 1) {
    if (t < s) red[t] += red[t + s];
    __syncthreads();
  }
  const float inv = 1.f / (red[0] + 1e-6f);
  att[(size_t)bk * GN + t] = e0 * inv;
  att[(size_t)bk * GN + 256 + t] = e1 * inv;
}

// ---------------------------------------------------------------------------
// Fused master-node path for one layer (tiny: 64 rows).  grid = B, block = 128.
// ---------------------------------------------------------------------------
__global__ __launch_bounds__(128)
void k_master(const float* __restrict__ att, const float* __restrict__ vf32,
              const float* __restrict__ master_in,
              const float* __restrict__ wmain_w, const float* __restrict__ wmain_b,
              const float* __restrict__ khead_w, const float* __restrict__ khead_b,
              const float* __restrict__ wmaster_w, const float* __restrict__ wmaster_b,
              const float* __restrict__ wm2m_w, const float* __restrict__ wm2m_b,
              const float* __restrict__ wzs1_w, const float* __restrict__ wzs1_b,
              const float* __restrict__ wzs2_w, const float* __restrict__ wzs2_b,
              const float* __restrict__ wzm2_w, const float* __restrict__ wzm2_b,
              const float* __restrict__ g_wih, const float* __restrict__ g_whh,
              const float* __restrict__ g_bih, const float* __restrict__ g_bhh,
              float* __restrict__ master_out, float* __restrict__ m2m_out,
              float* __restrict__ mterm_out)
{
  const int b = blockIdx.x, h = threadIdx.x;
  __shared__ float sS[GK][GH], sKmm[GK][GH], sSig[GK];
  __shared__ float sM2m[GH], sMself[GH], sMtm[GH], sHsup[GH], sMold[GH];
  __shared__ float sGi[3 * GH], sGh[3 * GH];

  sMold[h] = master_in[b * GH + h];
  for (int k = 0; k < GK; ++k) {
    float a = 0.f;
    for (int n = 0; n < GN; ++n)
      a += att[((size_t)b * GK + k) * GN + n] * vf32[((size_t)b * GN + n) * GH + h];
    sS[k][h] = a;
  }
  if (h < GK) {
    float a = 0.f;
    for (int n = 0; n < GN; ++n) a += att[((size_t)b * GK + h) * GN + n];
    sSig[h] = a;
  }
  __syncthreads();
  for (int k = 0; k < GK; ++k) {
    float a = 0.f;
    for (int j = 0; j < GH; ++j) a += sS[k][j] * wmain_w[((size_t)k * GH + j) * GH + h];
    sKmm[k][h] = a + sSig[k] * wmain_b[k * GH + h];
  }
  __syncthreads();
  {
    float a = 0.f;
    for (int idx = 0; idx < GK * GH; ++idx)
      a += sKmm[idx >> 7][idx & 127] * khead_w[(size_t)idx * GH + h];
    sMtm[h] = tanhf(a + khead_b[h]);
  }
  {
    float a = 0.f, c = 0.f;
    for (int j = 0; j < GH; ++j) {
      a += sMold[j] * wmaster_w[(size_t)j * GH + h];
      c += sMold[j] * wm2m_w[(size_t)j * GH + h];
    }
    sMself[h] = gelu_f(a + wmaster_b[h]);
    sM2m[h]   = gelu_f(c + wm2m_b[h]);
    m2m_out[b * GH + h] = sM2m[h];
  }
  __syncthreads();
  {
    float a = 0.f;
    for (int j = 0; j < GH; ++j) a += sM2m[j] * wzm2_w[(size_t)j * GH + h];
    mterm_out[b * GH + h] = a + wzm2_b[h];
  }
  {
    float a1 = 0.f, a2 = 0.f;
    for (int j = 0; j < GH; ++j) {
      a1 += sMself[j] * wzs1_w[(size_t)j * GH + h];
      a2 += sMtm[j]   * wzs2_w[(size_t)j * GH + h];
    }
    const float z = sigm_f(a1 + wzs1_b[h] + a2 + wzs2_b[h]);
    sHsup[h] = (1.f - z) * sMself[h] + z * sMtm[h];
  }
  __syncthreads();
  for (int t = 0; t < 3; ++t) {
    const int j = h + t * GH;
    float gia = 0.f, gha = 0.f;
    for (int k2 = 0; k2 < GH; ++k2) {
      gia += sHsup[k2] * g_wih[(size_t)j * GH + k2];
      gha += sMold[k2] * g_whh[(size_t)j * GH + k2];
    }
    sGi[j] = gia + g_bih[j];
    sGh[j] = gha + g_bhh[j];
  }
  __syncthreads();
  {
    const float r  = sigm_f(sGi[h] + sGh[h]);
    const float z2 = sigm_f(sGi[GH + h] + sGh[GH + h]);
    const float nn = tanhf(sGi[2 * GH + h] + r * sGh[2 * GH + h]);
    master_out[b * GH + h] = (1.f - z2) * nn + z2 * sMold[h];
  }
}

// ------------------------- small elementwise / prep kernels ------------------
__global__ void k_wt(const float* __restrict__ src, bf16_t* __restrict__ dst,
                     int K, int N, int Kp, int total) {
  int i = blockIdx.x * blockDim.x + threadIdx.x;
  if (i >= total) return;
  const int m = i / (N * Kp), r = i % (N * Kp), n = r / Kp, k = r % Kp;
  dst[i] = (k < K) ? (bf16_t)src[((size_t)m * K + k) * N + n] : (bf16_t)0.f;
}
__global__ void k_cvt(const float* __restrict__ s, bf16_t* __restrict__ d, int n) {
  int i = blockIdx.x * blockDim.x + threadIdx.x;
  if (i < n) d[i] = (bf16_t)s[i];
}
__global__ void k_copy32(float* __restrict__ d, const float* __restrict__ s, int n) {
  int i = blockIdx.x * blockDim.x + threadIdx.x;
  if (i < n) d[i] = s[i];
}
__global__ void k_eatom(const int* __restrict__ vertex, const float* __restrict__ vm,
                        const float* __restrict__ feat, bf16_t* __restrict__ Aemb,
                        int total) {
  int i = blockIdx.x * blockDim.x + threadIdx.x;
  if (i >= total) return;
  const int m = i / 96, c = i % 96;
  const float v = (c < 82) ? feat[vertex[m] * 82 + c] * vm[m] : 0.f;
  Aemb[i] = (bf16_t)v;
}
__global__ void k_ebond(const int* __restrict__ edge, const float* __restrict__ feat,
                        bf16_t* __restrict__ e16, int total) {
  int i = blockIdx.x * blockDim.x + threadIdx.x;
  if (i >= total) return;
  const int m = i / 8, c = i % 8;
  e16[i] = (c < 6) ? (bf16_t)feat[edge[m] * 6 + c] : (bf16_t)0.f;
}
__global__ void k_master_init(const float* __restrict__ vf, const float* __restrict__ vm,
                              float* __restrict__ mast) {
  const int b = blockIdx.x, h = threadIdx.x;
  float a = 0.f;
  for (int n = 0; n < GN; ++n)
    a += vf[((size_t)b * GN + n) * GH + h] * vm[b * GN + n];
  mast[b * GH + h] = a;
}
__global__ void k_gru_combine(const float* __restrict__ gi, const float* __restrict__ gh,
                              float* __restrict__ vf32, bf16_t* __restrict__ vf16, int n) {
  int i = blockIdx.x * blockDim.x + threadIdx.x;
  if (i >= n) return;
  const int m = i >> 7, h = i & 127;
  const float* gim = gi + (size_t)m * 384;
  const float* ghm = gh + (size_t)m * 384;
  const float r  = sigm_f(gim[h] + ghm[h]);
  const float z  = sigm_f(gim[GH + h] + ghm[GH + h]);
  const float nn = tanhf(gim[2 * GH + h] + r * ghm[2 * GH + h]);
  const float v = (1.f - z) * nn + z * vf32[i];
  vf32[i] = v;
  vf16[i] = (bf16_t)v;
}

// ---------------------------------------------------------------------------
extern "C" void kernel_launch(void* const* d_in, const int* in_sizes, int n_in,
                              void* d_out, int out_size, void* d_ws, size_t ws_size,
                              hipStream_t stream) {
  (void)in_sizes; (void)n_in; (void)out_size; (void)ws_size;
  const float* vm      = (const float*)d_in[1];
  const int*   vertex  = (const int*)d_in[2];
  const int*   edge    = (const int*)d_in[3];
  const int*   aa      = (const int*)d_in[4];
  const int*   ba      = (const int*)d_in[5];
  const float* nm      = (const float*)d_in[6];
  const float* feat_a  = (const float*)d_in[7];
  const float* feat_b  = (const float*)d_in[8];
  const float* emb_w   = (const float*)d_in[9];
  const float* emb_b   = (const float*)d_in[10];
  const float* u1_w    = (const float*)d_in[11];
  const float* u1_b    = (const float*)d_in[12];
  const float* u2_w    = (const float*)d_in[13];
  const float* u2_b    = (const float*)d_in[14];
  const float* fu_w    = (const float*)d_in[15];
  const float* fu_b    = (const float*)d_in[16];
  const float* wvm_w   = (const float*)d_in[17];
  const float* wvm_b   = (const float*)d_in[18];
  const float* wmain_w = (const float*)d_in[19];
  const float* wmain_b = (const float*)d_in[20];
  const float* wbmm_w  = (const float*)d_in[21];
  const float* wbmm_b  = (const float*)d_in[22];
  const float* khead_w = (const float*)d_in[23];
  const float* khead_b = (const float*)d_in[24];
  const float* wmaster_w = (const float*)d_in[25];
  const float* wmaster_b = (const float*)d_in[26];
  const float* wm2m_w  = (const float*)d_in[27];
  const float* wm2m_b  = (const float*)d_in[28];
  const float* wzm1_w  = (const float*)d_in[29];
  const float* wzm1_b  = (const float*)d_in[30];
  const float* wzm2_w  = (const float*)d_in[31];
  const float* wzm2_b  = (const float*)d_in[32];
  const float* wzs1_w  = (const float*)d_in[33];
  const float* wzs1_b  = (const float*)d_in[34];
  const float* wzs2_w  = (const float*)d_in[35];
  const float* wzs2_b  = (const float*)d_in[36];
  const float* gm_wih  = (const float*)d_in[37];
  const float* gm_whh  = (const float*)d_in[38];
  const float* gm_bih  = (const float*)d_in[39];
  const float* gm_bhh  = (const float*)d_in[40];
  const float* gs_wih  = (const float*)d_in[41];
  const float* gs_whh  = (const float*)d_in[42];
  const float* gs_bih  = (const float*)d_in[43];
  const float* gs_bhh  = (const float*)d_in[44];

  char* base = (char*)d_ws;
  size_t off = 0;
  auto alloc = [&](size_t bytes) -> char* {
    char* p = base + off;
    off = (off + bytes + 255) & ~(size_t)255;
    return p;
  };

  // weight prep buffers (bf16)
  bf16_t* embt  = (bf16_t*)alloc((size_t)128 * 96 * 2);
  bf16_t* u1t   = (bf16_t*)alloc((size_t)GL * 128 * 160 * 2);
  bf16_t* u2t   = (bf16_t*)alloc((size_t)GL * 128 * 256 * 2);
  bf16_t* fut   = (bf16_t*)alloc((size_t)GL * 128 * 128 * 2);
  bf16_t* wzm1t = (bf16_t*)alloc((size_t)GL * 128 * 128 * 2);
  bf16_t* wvmt  = (bf16_t*)alloc((size_t)GL * GK * 128 * 128 * 2);
  bf16_t* gih16 = (bf16_t*)alloc((size_t)384 * 128 * 2);
  bf16_t* ghh16 = (bf16_t*)alloc((size_t)384 * 128 * 2);
  // activations
  bf16_t* Aemb  = (bf16_t*)alloc((size_t)GM * 96 * 2);
  bf16_t* e16   = (bf16_t*)alloc((size_t)GB * GE * 8 * 2);
  float*  vf32  = (float*)alloc((size_t)GM * GH * 4);
  bf16_t* vf16  = (bf16_t*)alloc((size_t)GM * GH * 2);
  float*  h0    = (float*)alloc((size_t)GM * GH * 4);
  bf16_t* cat2  = (bf16_t*)alloc((size_t)GM * 256 * 2);
  float*  hi32  = (float*)alloc((size_t)GM * GH * 4);   // support
  float*  sv32  = (float*)alloc((size_t)GM * GH * 4);
  bf16_t* sv16  = (bf16_t*)alloc((size_t)GM * GH * 2);
  bf16_t* s16b  = (bf16_t*)alloc((size_t)GM * GH * 2);  // support bf16
  bf16_t* hid16 = (bf16_t*)alloc((size_t)GM * GH * 2);
  float*  gi32  = (float*)alloc((size_t)GM * 384 * 4);
  float*  gh32  = (float*)alloc((size_t)GM * 384 * 4);
  float*  score = (float*)alloc((size_t)GB * GK * GN * 4);
  float*  att   = (float*)alloc((size_t)GB * GK * GN * 4);
  float*  mastA = (float*)alloc((size_t)GB * GH * 4);
  float*  mastB = (float*)alloc((size_t)GB * GH * 4);
  float*  m2mB  = (float*)alloc((size_t)GB * GH * 4);
  float*  mterm = (float*)alloc((size_t)GB * GH * 4);

  const int T = 256;
  auto cdiv = [](int a, int b) { return (a + b - 1) / b; };
  auto lds  = [](int K) { return (size_t)128 * (K + 8) * 2; };

  // ---- weight prep ----
  k_wt<<<cdiv(128 * 96, T), T, 0, stream>>>(emb_w, embt, 82, 128, 96, 128 * 96);
  k_wt<<<cdiv(GL * 128 * 160, T), T, 0, stream>>>(u1_w, u1t, 134, 128, 160, GL * 128 * 160);
  k_wt<<<cdiv(GL * 128 * 256, T), T, 0, stream>>>(u2_w, u2t, 256, 128, 256, GL * 128 * 256);
  k_wt<<<cdiv(GL * 128 * 128, T), T, 0, stream>>>(fu_w, fut, 128, 128, 128, GL * 128 * 128);
  k_wt<<<cdiv(GL * 128 * 128, T), T, 0, stream>>>(wzm1_w, wzm1t, 128, 128, 128, GL * 128 * 128);
  k_wt<<<cdiv(GL * GK * 128 * 128, T), T, 0, stream>>>(wvm_w, wvmt, 128, 128, 128, GL * GK * 128 * 128);
  k_cvt<<<cdiv(384 * 128, T), T, 0, stream>>>(gm_wih, gih16, 384 * 128);
  k_cvt<<<cdiv(384 * 128, T), T, 0, stream>>>(gm_whh, ghh16, 384 * 128);

  // ---- embeddings ----
  k_eatom<<<cdiv(GM * 96, T), T, 0, stream>>>(vertex, vm, feat_a, Aemb, GM * 96);
  k_ebond<<<cdiv(GB * GE * 8, T), T, 0, stream>>>(edge, feat_b, e16, GB * GE * 8);
  k_gemm<96><<<dim3(GM / 128, 1), T, lds(96), stream>>>(Aemb, 96, embt, 96, emb_b,
      vf32, vf16, 128, nullptr, 0, /*gelu*/1, 0, nullptr, nullptr, 0.f, 0.f);
  k_copy32<<<cdiv(GM * GH, T), T, 0, stream>>>(h0, vf32, GM * GH);
  k_master_init<<<GB, GH, 0, stream>>>(vf32, vm, mastA);

  float* mCur = mastA;
  float* mNxt = mastB;
  for (int i = 0; i < GL; ++i) {
    const float theta = logf(LAMDA_C / (float)(i + 1) + 1.f);
    // attention scores + softmax
    k_score<<<dim3(GN / 128, GK, GB), T, 0, stream>>>(
        vf16, wvmt + (size_t)i * GK * GH * GH, wvm_b + (size_t)i * GK * GH,
        mCur, wbmm_w + (size_t)i * GK * GH, wbmm_b + (size_t)i * GK, score);
    k_softmax<<<GB * GK, T, 0, stream>>>(score, vm, att);
    // master path (uses pre-update vf/master; produces m2m, mterm, new master)
    k_master<<<GB, GH, 0, stream>>>(att, vf32, mCur,
        wmain_w + (size_t)i * GK * GH * GH, wmain_b + (size_t)i * GK * GH,
        khead_w + (size_t)i * GK * GH * GH, khead_b + (size_t)i * GH,
        wmaster_w + (size_t)i * GH * GH, wmaster_b + (size_t)i * GH,
        wm2m_w + (size_t)i * GH * GH, wm2m_b + (size_t)i * GH,
        wzs1_w + (size_t)i * GH * GH, wzs1_b + (size_t)i * GH,
        wzs2_w + (size_t)i * GH * GH, wzs2_b + (size_t)i * GH,
        wzm2_w + (size_t)i * GH * GH, wzm2_b + (size_t)i * GH,
        gs_wih, gs_whh, gs_bih, gs_bhh, mNxt, m2mB, mterm);
    // graph convolution (also writes vf16 into cat2[:,128:256])
    k_nbr<<<(GM * GNBS) / 128, T, 0, stream>>>(vf16, e16, aa, ba, nm,
        u1t + (size_t)i * 128 * 160, u1_b + (size_t)i * GH, cat2);
    // u2 GEMM with fused support: hi32 = (1-a)*(cat2@u2+b) + a*h0 ; s16b = bf16
    k_gemm<256><<<dim3(GM / 128, 1), T, lds(256), stream>>>(cat2, 256,
        u2t + (size_t)i * 128 * 256, 256, u2_b + (size_t)i * GH,
        hi32, s16b, 128, nullptr, 0, 0, 0, h0, nullptr,
        1.f - ALPHA_C, ALPHA_C);
    // fu GEMM: sv = theta*(support@fu + b) + (1-theta)*support
    k_gemm<128><<<dim3(GM / 128, 1), T, lds(128), stream>>>(s16b, 128,
        fut + (size_t)i * GH * GH, 128, fu_b + (size_t)i * GH,
        sv32, sv16, 128, nullptr, 0, 0, 0, hi32, nullptr,
        theta, 1.f - theta);
    // z_main GEMM with fused gated mixing:
    // z = sigmoid(sv@wzm1 + b + mterm[b]); hid16 = (1-z)*sv + z*m2m[b]
    k_gemm<128><<<dim3(GM / 128, 1), T, lds(128), stream>>>(sv16, 128,
        wzm1t + (size_t)i * GH * GH, 128, wzm1_b + (size_t)i * GH,
        nullptr, hid16, 128, mterm, 9, /*sigmoid*/2, /*mode*/1, sv32, m2mB,
        0.f, 0.f);
    // GRU main gates
    k_gemm<128><<<dim3(GM / 128, 3), T, lds(128), stream>>>(hid16, 128, gih16, 128,
        gm_bih, gi32, nullptr, 384, nullptr, 0, 0, 0, nullptr, nullptr,
        0.f, 0.f);
    k_gemm<128><<<dim3(GM / 128, 3), T, lds(128), stream>>>(vf16, 128, ghh16, 128,
        gm_bhh, gh32, nullptr, 384, nullptr, 0, 0, 0, nullptr, nullptr,
        0.f, 0.f);
    k_gru_combine<<<cdiv(GM * GH, T), T, 0, stream>>>(gi32, gh32, vf32, vf16, GM * GH);
    // swap master buffers
    float* tmp = mCur; mCur = mNxt; mNxt = tmp;
  }

  float* out = (float*)d_out;
  k_copy32<<<cdiv(GM * GH, T), T, 0, stream>>>(out, vf32, GM * GH);
  k_copy32<<<cdiv(GB * GH, T), T, 0, stream>>>(out + (size_t)GM * GH, mCur, GB * GH);
}
